// RBF_84859963834997
// MI455X (gfx1250) — matrix operand order repaired and checked by reading, not compiled, not run on hardware
//
#include <hip/hip_runtime.h>

typedef unsigned int u32;
typedef __attribute__((ext_vector_type(16))) _Float16 v16h;
typedef __attribute__((ext_vector_type(8)))  float    v8f;
typedef __attribute__((ext_vector_type(4)))  u32      v4u;
typedef __attribute__((ext_vector_type(8)))  int      v8i;
typedef __attribute__((ext_vector_type(4)))  int      v4i;

#define D_DIM   256
#define K_DIM   256
#define OUT_DIM 128

// ---------------------------------------------------------------------------
// Precompute 1: s = relu(sigma), w = relu(width), c2[k] = sum_d s_d*c[k,d]^2
// ---------------------------------------------------------------------------
__global__ void rbf_pre1(const float* __restrict__ sigma, const float* __restrict__ width,
                         const float* __restrict__ center,
                         float* __restrict__ s, float* __restrict__ w, float* __restrict__ c2) {
  __shared__ float sh_s[D_DIM];
  int t = threadIdx.x;                       // 0..255
  float sv = fmaxf(sigma[t], 0.0f);
  s[t] = sv;
  w[t] = fmaxf(width[t], 0.0f);
  sh_s[t] = sv;
  __syncthreads();
  const float* c = center + t * D_DIM;       // row t of center
  float acc = 0.0f;
#pragma unroll 4
  for (int d = 0; d < D_DIM; ++d) { float cv = c[d]; acc = fmaf(sh_s[d] * cv, cv, acc); }
  c2[t] = acc;
}

// ---------------------------------------------------------------------------
// Precompute 2: S1[o] = sum_k w_k*Wr[1+k,o];  bias[o] = Wr[0,o] + sum_k (1 - w_k*c2[k])*Wr[1+k,o]
// ---------------------------------------------------------------------------
__global__ void rbf_pre2(const float* __restrict__ W, const float* __restrict__ w,
                         const float* __restrict__ c2,
                         float* __restrict__ S1, float* __restrict__ bias) {
  int o = threadIdx.x;                       // 0..127
  float s1 = 0.0f, b = 0.0f;
  for (int k = 0; k < K_DIM; ++k) {
    float wr = fmaxf(W[(1 + k) * OUT_DIM + o], 0.0f);
    float wk = w[k];
    s1 = fmaf(wk, wr, s1);
    b  = fmaf(1.0f - wk * c2[k], wr, b);
  }
  S1[o]   = s1;
  bias[o] = fmaxf(W[o], 0.0f) + b;
}

// ---------------------------------------------------------------------------
// Precompute 3: M2[d,o] = 2*s_d * sum_k c[k,d]*w_k*Wr[1+k,o], stored as f16 in
// WMMA B-fragment order: frag (t = d/32, c = o/16), lane = (o&15) | (((d>>3)&1)<<4),
// 16 halfs per lane at element (d&7) | (((d>>4)&1)<<3).
// ---------------------------------------------------------------------------
__global__ void rbf_pre3(const float* __restrict__ center, const float* __restrict__ W,
                         const float* __restrict__ w, const float* __restrict__ s,
                         u32* __restrict__ M2h) {
  int d = blockIdx.x;                        // 0..255
  int o = threadIdx.x;                       // 0..127
  float acc = 0.0f;
  for (int k = 0; k < K_DIM; ++k)
    acc = fmaf(center[k * D_DIM + d] * w[k], fmaxf(W[(1 + k) * OUT_DIM + o], 0.0f), acc);
  float val = 2.0f * s[d] * acc;
  int t    = d >> 5;
  int c    = o >> 4;
  int lane = (o & 15) | (((d >> 3) & 1) << 4);
  int idx  = (((t * 8 + c) * 32) + lane) * 16 + ((d & 7) | (((d >> 4) & 1) << 3));
  reinterpret_cast<_Float16*>(M2h)[idx] = (_Float16)val;
}

// ---------------------------------------------------------------------------
// Main: out[n,o] = (f16 x) @ (f16 M2) [f32 acc WMMA] + bias[o] - x2[n]*S1[o]
// Block: 256 threads = 8 waves; 128 rows x 128 cols per block; K-loop 8x32.
// ---------------------------------------------------------------------------
__global__ void __launch_bounds__(256) rbf_main(
    const float* __restrict__ x, const u32* __restrict__ M2h,
    const float* __restrict__ s, const float* __restrict__ S1,
    const float* __restrict__ bias, float* __restrict__ out) {
  __shared__ u32 ldsM[16384];                // 64 KB of pre-swizzled f16 B fragments

  // ---- Stage M2 into LDS: TDM tensor DMA (wave 0 issues, TENSORcnt wait) ----
#if __has_builtin(__builtin_amdgcn_tensor_load_to_lds)
  if (threadIdx.x < 32) {
    u32 lds_addr = (u32)(size_t)(void*)&ldsM[0];
    unsigned long long ga = (unsigned long long)(size_t)M2h;
    v4u g0; g0[0] = 1u;                       // count=1, no gather
    g0[1] = lds_addr;                          // lds_addr [63:32]
    g0[2] = (u32)ga;                           // global_addr lo
    g0[3] = ((u32)(ga >> 32) & 0x1FFFFFFu) | (2u << 30);  // global_addr hi | type=2
    v8i g1;
    g1[0] = (int)(3u << 16);                   // workgroup_mask=0, data_size=8B
    g1[1] = (int)(0x2000u << 16);              // tensor_dim0 = 8192 (lo 16 in [31:16])
    g1[2] = (int)(1u << 16);                   // tensor_dim0 hi=0, tensor_dim1=1
    g1[3] = (int)(0x2000u << 16);              // tensor_dim1 hi=0, tile_dim0=8192
    g1[4] = 1;                                 // tile_dim1=1, tile_dim2=0
    g1[5] = 8192;                              // tensor_dim0_stride lo
    g1[6] = 0; g1[7] = 0;
    v4i gz4; gz4[0] = 0; gz4[1] = 0; gz4[2] = 0; gz4[3] = 0;
    v8i gz8; gz8[0] = 0; gz8[1] = 0; gz8[2] = 0; gz8[3] = 0;
            gz8[4] = 0; gz8[5] = 0; gz8[6] = 0; gz8[7] = 0;
    // 6-arg form (clang-23 / therock headers): (g0, g1, g2, g3, g4, cpol)
    __builtin_amdgcn_tensor_load_to_lds(g0, g1, gz4, gz4, gz8, 0);
    __builtin_amdgcn_s_wait_tensorcnt(0);
  }
#else
  {
    const uint4* src = reinterpret_cast<const uint4*>(M2h);
    uint4* dst = reinterpret_cast<uint4*>(ldsM);
    for (int i = threadIdx.x; i < 4096; i += 256) dst[i] = src[i];
  }
#endif
  __syncthreads();

  const int lane = threadIdx.x & 31;
  const int wv   = threadIdx.x >> 5;
  const int r0   = blockIdx.x * 128 + wv * 16;            // 16-row tile of this wave
  const int mrow = lane & 15;
  const int koff = (lane < 16) ? 0 : 8;                   // A-frag lane-half K split
  const float* xrow = x + (size_t)(r0 + mrow) * D_DIM;

  v8f acc[8];
#pragma unroll
  for (int c = 0; c < 8; ++c) { v8f z = {0.f,0.f,0.f,0.f,0.f,0.f,0.f,0.f}; acc[c] = z; }

  float x2p = 0.0f;                                       // partial sum_d s_d x_d^2

#pragma unroll
  for (int t = 0; t < 8; ++t) {
    const int k0 = t * 32 + koff;
    float4 a0 = *reinterpret_cast<const float4*>(xrow + k0);
    float4 a1 = *reinterpret_cast<const float4*>(xrow + k0 + 4);
    float4 a2 = *reinterpret_cast<const float4*>(xrow + k0 + 16);
    float4 a3 = *reinterpret_cast<const float4*>(xrow + k0 + 20);
    float av[16] = {a0.x,a0.y,a0.z,a0.w, a1.x,a1.y,a1.z,a1.w,
                    a2.x,a2.y,a2.z,a2.w, a3.x,a3.y,a3.z,a3.w};
#pragma unroll
    for (int j = 0; j < 8; ++j) x2p = fmaf(s[k0 + j]      * av[j],     av[j],     x2p);
#pragma unroll
    for (int j = 0; j < 8; ++j) x2p = fmaf(s[k0 + 16 + j] * av[8 + j], av[8 + j], x2p);

    v16h A;                                                // ISA 16-bit A 16x32 layout
#pragma unroll
    for (int j = 0; j < 16; ++j) A[j] = (_Float16)av[j];

#pragma unroll
    for (int c = 0; c < 8; ++c) {                          // 8 column tiles of OUT
      const v16h B = *reinterpret_cast<const v16h*>(&ldsM[((t * 8 + c) * 32 + lane) * 8]);
      acc[c] = __builtin_amdgcn_wmma_f32_16x16x32_f16(
          false, A, false, B, (short)0, acc[c], false, false);
    }
  }

  // complete x2 for this lane's row: lane halves hold disjoint K subsets
  float x2 = x2p + __shfl_xor(x2p, 16, 32);

  // epilogue: C/D layout — VGPR j: lanes0-15 -> M=j, lanes16-31 -> M=j+8; N = lane%16
#pragma unroll
  for (int c = 0; c < 8; ++c) {
    const int col = c * 16 + mrow;
    const float s1v = S1[col];
    const float bv  = bias[col];
#pragma unroll
    for (int j = 0; j < 8; ++j) {
      const int rit  = j + ((lane < 16) ? 0 : 8);
      const float x2r = __shfl(x2, rit, 32);               // lane rit holds row rit
      out[(size_t)(r0 + rit) * OUT_DIM + col] = acc[c][j] + bv - x2r * s1v;
    }
  }
}

// ---------------------------------------------------------------------------
extern "C" void kernel_launch(void* const* d_in, const int* in_sizes, int n_in,
                              void* d_out, int out_size, void* d_ws, size_t ws_size,
                              hipStream_t stream) {
  const float* x      = (const float*)d_in[0];   // [N,256]
  const float* center = (const float*)d_in[1];   // [256,256]
  const float* sigma  = (const float*)d_in[2];   // [1,256]
  const float* width  = (const float*)d_in[3];   // [1,256]
  const float* W      = (const float*)d_in[4];   // [257,128]
  float* out = (float*)d_out;

  const int N = in_sizes[0] / D_DIM;             // 131072

  // workspace layout
  char*  ws   = (char*)d_ws;
  u32*   M2h  = (u32*)ws;                        // 65536 B: f16 B-fragments
  float* s    = (float*)(ws + 65536);            // 256
  float* w    = s + 256;                         // 256
  float* c2   = w + 256;                         // 256
  float* S1   = c2 + 256;                        // 128
  float* bias = S1 + 128;                        // 128

  rbf_pre1<<<1, 256, 0, stream>>>(sigma, width, center, s, w, c2);
  rbf_pre2<<<1, 128, 0, stream>>>(W, w, c2, S1, bias);
  rbf_pre3<<<256, 128, 0, stream>>>(center, W, w, s, M2h);
  rbf_main<<<N / 128, 256, 0, stream>>>(x, M2h, s, S1, bias, out);
}
